// MoEAdaptorLayer_111669150283
// MI455X (gfx1250) — matrix operand order repaired and checked by reading, not compiled, not run on hardware
//
#include <hip/hip_runtime.h>

// ---------------------------------------------------------------------------
// MoE adaptor layer for MI455X (gfx1250, wave32, WMMA).
// Dominant cost: token×(E*O) GEMM, M=25600, N=2400, K=768 (94.4 GFLOP).
// Strategy: bf16x3 split (hi/lo) + v_wmma_f32_16x16x32_bf16, expert reduction
// fused in registers, W chunks streamed with async global->LDS copies.
// ---------------------------------------------------------------------------

#ifndef __has_builtin
#define __has_builtin(x) 0
#endif

#if __has_builtin(__builtin_amdgcn_global_load_async_to_lds_b128)
#define HAVE_ASYNC_LDS 1
#else
#define HAVE_ASYNC_LDS 0
#endif

typedef __attribute__((ext_vector_type(16))) __bf16 v16bf;
typedef __attribute__((ext_vector_type(4)))  __bf16 v4bf;
typedef __attribute__((ext_vector_type(8)))  float  v8f;

// Exact parameter types for the async-LDS builtin (from compiler diagnostic):
// (v4i addrspace(1)*, v4i addrspace(3)*, imm offset, imm cpol)
typedef int v4i __attribute__((vector_size(16)));
typedef __attribute__((address_space(1))) v4i gv4i;
typedef __attribute__((address_space(3))) v4i lv4i;

union Frag {
  uint4 u[2];
  v16bf v;
};

// Problem constants
#define TOKENS 25600   // 512*50
#define DIN    768
#define DOUT   300
#define NEXP   8
#define SEQL   50
#define OPAD   384     // padded expert output dim for guard-free tile loads

// Main-kernel tiling
#define M_BLK  64
#define N_BLK  128
#define KCH    32
#define NKC    (DIN / KCH)      // 24
#define NCHUNK (NEXP * NKC)     // 192

// LDS layout (dynamic). X rows padded (+16B) for conflict-free ds_load_b128.
#define XROW     1552           // 768*2 + 16
#define XH_OFF   0
#define XL_OFF   (M_BLK * XROW)                   // 99328
#define W_OFF    (2 * M_BLK * XROW)               // 198656
#define WROW     80                                // 32*2 + 16 pad
#define WH_SZ    (N_BLK * WROW)                    // 10240
#define WBUF_SZ  (2 * WH_SZ)                       // hi+lo per buffer
#define G_OFF    (W_OFF + 2 * WBUF_SZ)             // 239616
#define SMEM_TOTAL (G_OFF + M_BLK * NEXP * 4)      // 241664 (< 320KB/WGP)

// Workspace layout (bytes)
#define WS_GATES 0
#define WS_BW    819200                     // 25600*8*4
#define WS_WH    1299200                    // + 50*8*300*4
#define WS_WL    6017792                    // + 8*384*768*2

__device__ __forceinline__ void async_copy_b128(const __bf16* gsrc, char* ldst) {
#if HAVE_ASYNC_LDS
  __builtin_amdgcn_global_load_async_to_lds_b128((gv4i*)gsrc, (lv4i*)ldst, 0, 0);
#else
  *(uint4*)ldst = *(const uint4*)gsrc;
#endif
}

__device__ __forceinline__ void wait_async0() {
#if HAVE_ASYNC_LDS
#if __has_builtin(__builtin_amdgcn_s_wait_asynccnt)
  __builtin_amdgcn_s_wait_asynccnt(0);
#else
  asm volatile("s_wait_asynccnt 0" ::: "memory");
#endif
#endif
}

// ---------------------------------------------------------------------------
// Kernel 1: gating softmax.  gates[t][e] = softmax_e(x[t,:] @ w_gate[:,e])
// One wave per token: lane = (quarter q = lane>>3, expert e = lane&7).
// ---------------------------------------------------------------------------
__global__ __launch_bounds__(256) void gates_kernel(
    const float* __restrict__ x, const float* __restrict__ wg,
    float* __restrict__ gates) {
  const int wave = threadIdx.x >> 5;
  const int lane = threadIdx.x & 31;
  const int token = blockIdx.x * 8 + wave;
  const int e = lane & 7;
  const int q = lane >> 3;

  const float* xr = x + (size_t)token * DIN + q * 192;
  const float* wr = wg + (size_t)q * 192 * NEXP + e;
  float acc = 0.f;
#pragma unroll 4
  for (int j = 0; j < 192; ++j) acc += xr[j] * wr[j * NEXP];
  // reduce the 4 quarters (lanes xor 8, 16)
  acc += __shfl_xor(acc, 8);
  acc += __shfl_xor(acc, 16);
  // softmax across the 8 experts within each group of 8 lanes
  float m = acc;
  m = fmaxf(m, __shfl_xor(m, 1));
  m = fmaxf(m, __shfl_xor(m, 2));
  m = fmaxf(m, __shfl_xor(m, 4));
  float ev = __expf(acc - m);
  float s = ev;
  s += __shfl_xor(s, 1);
  s += __shfl_xor(s, 2);
  s += __shfl_xor(s, 4);
  if (lane < 8) gates[(size_t)token * NEXP + e] = ev / s;
}

// ---------------------------------------------------------------------------
// Kernel 2: bw[l][e][o] = sum_i bias[e,l,i] * W[e,o,i]   (tiny, 0.18 GFLOP)
// blockIdx = l*8+e ; bias row staged in LDS.
// ---------------------------------------------------------------------------
__global__ __launch_bounds__(320) void bw_kernel(
    const float* __restrict__ expert_bias, const float* __restrict__ expert_w,
    float* __restrict__ bw) {
  const int l = blockIdx.x >> 3;
  const int e = blockIdx.x & 7;
  __shared__ float brow[DIN];
  for (int i = threadIdx.x; i < DIN; i += blockDim.x)
    brow[i] = expert_bias[((size_t)e * SEQL + l) * DIN + i];
  __syncthreads();
  const int o = threadIdx.x;
  if (o < DOUT) {
    const float* wr = expert_w + ((size_t)e * DOUT + o) * DIN;
    float acc = 0.f;
#pragma unroll 4
    for (int i = 0; i < DIN; i += 4) {
      float4 w4 = *(const float4*)(wr + i);
      acc += brow[i] * w4.x + brow[i + 1] * w4.y + brow[i + 2] * w4.z +
             brow[i + 3] * w4.w;
    }
    bw[(size_t)l * (NEXP * DOUT) + e * DOUT + o] = acc;
  }
}

// ---------------------------------------------------------------------------
// Kernel 3: split expert_w (f32) into bf16 hi/lo, O padded 300 -> 384.
// ---------------------------------------------------------------------------
__global__ __launch_bounds__(256) void wsplit_kernel(
    const float* __restrict__ W, __bf16* __restrict__ WH,
    __bf16* __restrict__ WL) {
  const size_t idx = (size_t)blockIdx.x * 256 + threadIdx.x;  // < 8*384*768
  const int i = (int)(idx % DIN);
  const int o = (int)((idx / DIN) % OPAD);
  const int e = (int)(idx / ((size_t)DIN * OPAD));
  float v = (o < DOUT) ? W[((size_t)e * DOUT + o) * DIN + i] : 0.f;
  __bf16 h = (__bf16)v;
  __bf16 lo = (__bf16)(v - (float)h);
  WH[idx] = h;
  WL[idx] = lo;
}

// ---------------------------------------------------------------------------
// Kernel 4: main fused GEMM.
//   grid.x = 400 (64-token tiles), grid.y = 3 (128-output tiles)
//   8 waves; wave (waveM 0..1, waveN 0..3) owns a 32x32 sub-tile (2x2 frags).
//   X tile (64x768) converted to bf16 hi/lo once, resident in LDS.
//   Expert loop streams W hi/lo k-chunks via double-buffered async copies.
//   out[t][o] = sum_e g[t,e]*(x[t]·W[e,o]) - sum_e g[t,e]*bw[l,e,o]
// ---------------------------------------------------------------------------
__global__ __launch_bounds__(256) void moe_main_kernel(
    const float* __restrict__ x, const __bf16* __restrict__ WH,
    const __bf16* __restrict__ WL, const float* __restrict__ gates,
    const float* __restrict__ bw, float* __restrict__ out) {
  extern __shared__ char smem[];
  const int tid = threadIdx.x;
  const int lane = tid & 31;
  const int wave = tid >> 5;
  const int waveM = wave >> 2;  // 0..1
  const int waveN = wave & 3;   // 0..3
  const int g16 = lane >> 4;    // lane half-group (WMMA layout)
  const int l16 = lane & 15;
  const int mBase = blockIdx.x * M_BLK;
  const int nBase = blockIdx.y * N_BLK;

  // ---- Stage X tile: load f32, split to bf16 hi/lo in LDS ----
  {
    const float4* xsrc = (const float4*)(x + (size_t)mBase * DIN);
#pragma unroll 4
    for (int it = 0; it < 48; ++it) {
      int idx = it * 256 + tid;  // 0..12287 over 64 rows x 192 float4
      int row = idx / 192;
      int kq = idx % 192;
      float4 v = xsrc[(size_t)row * 192 + kq];
      v4bf h, lo;
      h[0] = (__bf16)v.x; h[1] = (__bf16)v.y;
      h[2] = (__bf16)v.z; h[3] = (__bf16)v.w;
      lo[0] = (__bf16)(v.x - (float)h[0]);
      lo[1] = (__bf16)(v.y - (float)h[1]);
      lo[2] = (__bf16)(v.z - (float)h[2]);
      lo[3] = (__bf16)(v.w - (float)h[3]);
      *(v4bf*)(smem + XH_OFF + (size_t)row * XROW + kq * 8) = h;
      *(v4bf*)(smem + XL_OFF + (size_t)row * XROW + kq * 8) = lo;
    }
    // Stage gates for the 64 tokens (512 floats)
    if (tid < 128) {
      float4 gv = *(const float4*)(gates + (size_t)mBase * NEXP + tid * 4);
      *(float4*)(smem + G_OFF + tid * 16) = gv;
    }
  }

  // W chunk staging: 128 rows(o) x 32 k of bf16 hi+lo -> LDS [n][k] rows.
  auto issueChunk = [&](int c, int b) {
    const int e = c / NKC;
    const int kc = c - e * NKC;
    const size_t srcBase =
        ((size_t)e * OPAD + nBase) * DIN + (size_t)kc * KCH;  // element offset
    char* wb = smem + W_OFF + b * WBUF_SZ;
#pragma unroll
    for (int half = 0; half < 2; ++half) {
      const int n = half * 64 + (tid >> 2);
      const int q = tid & 3;
      const size_t off = srcBase + (size_t)n * DIN + q * 8;
      char* d = wb + n * WROW + q * 16;
      async_copy_b128(WH + off, d);          // hi
      async_copy_b128(WL + off, d + WH_SZ);  // lo
    }
  };

  v8f tacc[2][2] = {};  // per-expert temp accumulators
  v8f oacc[2][2] = {};  // gate-weighted output accumulators

  issueChunk(0, 0);

  for (int c = 0; c < NCHUNK; ++c) {
    const int b = c & 1;
    wait_async0();     // my async writes for chunk c are in LDS
    __syncthreads();   // everyone's writes visible; prev-buffer reads retired
    if (c + 1 < NCHUNK) issueChunk(c + 1, b ^ 1);  // overlap copy w/ compute

    const int kc = c % NKC;

    // A fragments from resident X LDS (layout per ISA 16-bit A 16x32)
    Frag AH[2], AL[2];
#pragma unroll
    for (int mf = 0; mf < 2; ++mf) {
      const int row = waveM * 32 + mf * 16 + l16;
      const size_t ao = (size_t)row * XROW + (size_t)kc * 64 + g16 * 16;
      AH[mf].u[0] = *(const uint4*)(smem + XH_OFF + ao);
      AH[mf].u[1] = *(const uint4*)(smem + XH_OFF + ao + 32);
      AL[mf].u[0] = *(const uint4*)(smem + XL_OFF + ao);
      AL[mf].u[1] = *(const uint4*)(smem + XL_OFF + ao + 32);
    }
    // B fragments from W chunk buffer (layout per ISA 16-bit B 32x16)
    Frag BH[2], BL[2];
    char* wb = smem + W_OFF + b * WBUF_SZ;
#pragma unroll
    for (int nf = 0; nf < 2; ++nf) {
      const int col = waveN * 32 + nf * 16 + l16;
      const size_t bo = (size_t)col * WROW + g16 * 16;
      BH[nf].u[0] = *(const uint4*)(wb + bo);
      BH[nf].u[1] = *(const uint4*)(wb + bo + 32);
      BL[nf].u[0] = *(const uint4*)(wb + WH_SZ + bo);
      BL[nf].u[1] = *(const uint4*)(wb + WH_SZ + bo + 32);
    }

    // bf16x3: hi*hi + hi*lo + lo*hi, f32 accumulate
#pragma unroll
    for (int mf = 0; mf < 2; ++mf)
#pragma unroll
      for (int nf = 0; nf < 2; ++nf) {
        v8f t = tacc[mf][nf];
        t = __builtin_amdgcn_wmma_f32_16x16x32_bf16(false, AH[mf].v, false,
                                                    BH[nf].v, (short)0, t,
                                                    false, false);
        t = __builtin_amdgcn_wmma_f32_16x16x32_bf16(false, AH[mf].v, false,
                                                    BL[nf].v, (short)0, t,
                                                    false, false);
        t = __builtin_amdgcn_wmma_f32_16x16x32_bf16(false, AL[mf].v, false,
                                                    BH[nf].v, (short)0, t,
                                                    false, false);
        tacc[mf][nf] = t;
      }

    // Expert boundary: fold gate-weighted temp into output accumulator
    if (kc == NKC - 1) {
      const int e = c / NKC;
#pragma unroll
      for (int mf = 0; mf < 2; ++mf) {
        float gv[8];
#pragma unroll
        for (int r = 0; r < 8; ++r) {
          const int row = waveM * 32 + mf * 16 + g16 * 8 + r;  // C/D layout
          gv[r] = *(const float*)(smem + G_OFF + (row * NEXP + e) * 4);
        }
#pragma unroll
        for (int nf = 0; nf < 2; ++nf)
#pragma unroll
          for (int r = 0; r < 8; ++r) {
            oacc[mf][nf][r] += gv[r] * tacc[mf][nf][r];
            tacc[mf][nf][r] = 0.f;
          }
      }
    }
  }

  // ---- Epilogue: subtract sum_e g*bw and store ----
#pragma unroll
  for (int mf = 0; mf < 2; ++mf) {
    const int rowBase = waveM * 32 + mf * 16 + g16 * 8;
#pragma unroll
    for (int nf = 0; nf < 2; ++nf) {
      const int col = nBase + waveN * 32 + nf * 16 + l16;
      if (col < DOUT) {
#pragma unroll
        for (int r = 0; r < 8; ++r) {
          const int row = rowBase + r;
          const int token = mBase + row;
          const int l = token % SEQL;
          const float* gp = (const float*)(smem + G_OFF) + row * NEXP;
          const float* bwp = bw + (size_t)l * (NEXP * DOUT) + col;
          float corr = 0.f;
#pragma unroll
          for (int e = 0; e < NEXP; ++e) corr += gp[e] * bwp[e * DOUT];
          out[(size_t)token * DOUT + col] = oacc[mf][nf][r] - corr;
        }
      }
    }
  }
}

// ---------------------------------------------------------------------------
extern "C" void kernel_launch(void* const* d_in, const int* in_sizes, int n_in,
                              void* d_out, int out_size, void* d_ws,
                              size_t ws_size, hipStream_t stream) {
  const float* x  = (const float*)d_in[0];   // [512,50,768]
  const float* wg = (const float*)d_in[1];   // [768,8]
  const float* ew = (const float*)d_in[2];   // [8,300,768]
  const float* eb = (const float*)d_in[3];   // [8,50,768]
  float* out = (float*)d_out;                // [512,50,300]

  char* ws = (char*)d_ws;                    // needs ~10.8 MB
  float* gates = (float*)(ws + WS_GATES);
  float* bwbuf = (float*)(ws + WS_BW);
  __bf16* WH = (__bf16*)(ws + WS_WH);
  __bf16* WL = (__bf16*)(ws + WS_WL);

  gates_kernel<<<TOKENS / 8, 256, 0, stream>>>(x, wg, gates);
  bw_kernel<<<SEQL * NEXP, 320, 0, stream>>>(eb, ew, bwbuf);
  wsplit_kernel<<<(NEXP * OPAD * DIN) / 256, 256, 0, stream>>>(ew, WH, WL);

  (void)hipFuncSetAttribute((const void*)moe_main_kernel,
                            hipFuncAttributeMaxDynamicSharedMemorySize,
                            SMEM_TOTAL);
  dim3 grid(TOKENS / M_BLK, (DOUT + N_BLK - 1) / N_BLK);  // 400 x 3
  moe_main_kernel<<<grid, 256, SMEM_TOTAL, stream>>>(x, WH, WL, gates, bwbuf,
                                                     out);
}
